// AttentionBlock3D_48421461295757
// MI455X (gfx1250) — matrix operand order repaired and checked by reading, not compiled, not run on hardware
//
#include <hip/hip_runtime.h>
#include <hip/hip_bf16.h>
#include <stdint.h>

// ---------------- problem constants ----------------
#define B_   2
#define C_   256
#define N_   8192      // D*H*W = 8*32*32
#define G_   32
#define CPG  8         // channels per group
#define OQKV (3 * C_)  // 768

typedef __attribute__((ext_vector_type(16))) __bf16 v16bf;
typedef __attribute__((ext_vector_type(8)))  float  v8f;

struct U4 { uint32_t x, y, z, w; };

union AF {               // one 16x32 bf16 WMMA fragment (A or B), 8 VGPRs
    v16bf    v;
    uint32_t w[8];
};

__device__ __forceinline__ unsigned short f2bf(float f) {
    union { float f; uint32_t u; } c; c.f = f;
    uint32_t u = c.u;
    uint32_t r = u + 0x7FFFu + ((u >> 16) & 1u);   // round-to-nearest-even
    return (unsigned short)(r >> 16);
}
__device__ __forceinline__ uint32_t pack2bf(float lo, float hi) {
    return (uint32_t)f2bf(lo) | ((uint32_t)f2bf(hi) << 16);
}
__device__ __forceinline__ void load16B(uint32_t* dst, const void* src) {
    U4 t = *(const U4*)src;
    dst[0] = t.x; dst[1] = t.y; dst[2] = t.z; dst[3] = t.w;
}
__device__ __forceinline__ v8f wmma_bf16(v16bf a, v16bf b, v8f c) {
    return __builtin_amdgcn_wmma_f32_16x16x32_bf16(
        /*neg_a=*/false, a, /*neg_b=*/false, b,
        /*c_mod=*/(short)0, c, /*reuse_a=*/false, /*reuse_b=*/false);
}

// ---------------------------------------------------------------------------
// Kernel 0: one-shot weight conversion f32 -> bf16 (row-major planes).
// qkv_w [768][256] -> wq_bf; proj_w [256][256] -> wp_bf.  Done once so the
// GEMM inner loops load A fragments with two b128 loads instead of doing
// f32->bf16 packing per tile.
// ---------------------------------------------------------------------------
__global__ __launch_bounds__(256) void wconv_kernel(
    const float* __restrict__ qkv_w, const float* __restrict__ proj_w,
    uint32_t* __restrict__ wq_bf, uint32_t* __restrict__ wp_bf) {
    const int nq = (OQKV * C_) / 2;      // 98304 dwords
    const int np = (C_ * C_) / 2;        // 32768 dwords
    int i = blockIdx.x * blockDim.x + threadIdx.x;
    if (i < nq) {
        wq_bf[i] = pack2bf(qkv_w[2 * i], qkv_w[2 * i + 1]);
    } else if (i < nq + np) {
        int j = i - nq;
        wp_bf[j] = pack2bf(proj_w[2 * j], proj_w[2 * j + 1]);
    }
}

// ---------------------------------------------------------------------------
// Kernel 1: GroupNorm.  One block per (batch, group).  Writes bf16 pairs
// packed along channel:  h_pk[b][c/2][n] = (bf16 h[c][n], bf16 h[c+1][n]).
// ---------------------------------------------------------------------------
__global__ __launch_bounds__(256) void gn_kernel(
    const float* __restrict__ x, const float* __restrict__ gw,
    const float* __restrict__ gb, uint32_t* __restrict__ h_pk) {
    int bg = blockIdx.x;            // 0..63
    int b = bg / G_, g = bg % G_;
    const float* xb = x + ((size_t)b * C_ + (size_t)g * CPG) * N_;
    int tid = threadIdx.x;
    const int total = CPG * N_;

    float s = 0.f, ss = 0.f;
    for (int i = tid; i < total; i += 256) {
        float v = xb[i];
        s += v; ss += v * v;
    }
    __shared__ float rs[256], rss[256];
    rs[tid] = s; rss[tid] = ss;
    __syncthreads();
    for (int off = 128; off > 0; off >>= 1) {
        if (tid < off) { rs[tid] += rs[tid + off]; rss[tid] += rss[tid + off]; }
        __syncthreads();
    }
    float mean = rs[0] * (1.0f / total);
    float var  = rss[0] * (1.0f / total) - mean * mean;
    float rstd = rsqrtf(var + 1e-5f);

    uint32_t* hb = h_pk + ((size_t)b * (C_ / 2) + (size_t)g * (CPG / 2)) * N_;
    const int words = (CPG / 2) * N_;
    for (int i = tid; i < words; i += 256) {
        int c2 = i / N_;            // 0..3
        int n  = i - c2 * N_;
        int c  = g * CPG + 2 * c2;
        float v0 = (xb[(2 * c2)     * N_ + n] - mean) * rstd * gw[c]     + gb[c];
        float v1 = (xb[(2 * c2 + 1) * N_ + n] - mean) * rstd * gw[c + 1] + gb[c + 1];
        hb[i] = pack2bf(v0, v1);
    }
}

// ---------------------------------------------------------------------------
// Kernel 2: QKV = qkv_w @ h (+bias).  One wave per 16(o) x 16(n) tile,
// K-loop over C=256 in 32-steps with v_wmma_f32_16x16x32_bf16.
// Outputs:  qs_t  [B][N][C]   bf16, pre-scaled by C^-0.5   (A-frag layout)
//           k_pk  [B][C/2][N] dwords of (c,c+1) bf16 pairs (B-frag layout)
//           v_pk  [B][N/2][C] dwords of (n,n+1) bf16 pairs (B-frag layout)
// ---------------------------------------------------------------------------
__global__ __launch_bounds__(256) void qkv_kernel(
    const uint32_t* __restrict__ h_pk, const unsigned short* __restrict__ wq_bf,
    const float* __restrict__ qkv_b, unsigned short* __restrict__ qs_t,
    uint32_t* __restrict__ k_pk, uint32_t* __restrict__ v_pk) {
    int wave = (int)((blockIdx.x * blockDim.x + threadIdx.x) >> 5);
    int lane = threadIdx.x & 31;
    int nt = wave % (N_ / 16);
    int ot = (wave / (N_ / 16)) % (OQKV / 16);
    int b  = wave / ((N_ / 16) * (OQKV / 16));
    int o0 = ot * 16, n0 = nt * 16;
    int row = lane & 15;
    int hi  = lane >> 4;    // 0: lanes 0-15, 1: lanes 16-31

    const uint32_t* hb = h_pk + (size_t)b * (C_ / 2) * N_;
    const unsigned short* wrow = wq_bf + (size_t)(o0 + row) * C_;
    v8f acc = {0.f,0.f,0.f,0.f,0.f,0.f,0.f,0.f};

    for (int c0 = 0; c0 < C_; c0 += 32) {
        AF A, Bf;
        // A: bf16 weight row, two 16B loads
        const unsigned short* wp = wrow + c0 + hi * 16;
        load16B(&A.w[0], wp);
        load16B(&A.w[4], wp + 8);
        // B: h pairs along c at column n
        int cb = (c0 >> 1) + hi * 8;
#pragma unroll
        for (int vv = 0; vv < 8; ++vv)
            Bf.w[vv] = hb[(size_t)(cb + vv) * N_ + n0 + row];
        acc = wmma_bf16(A.v, Bf.v, acc);
    }
#pragma unroll
    for (int r = 0; r < 8; ++r) acc[r] += qkv_b[o0 + hi * 8 + r];

    int col = n0 + row;     // n column owned by this lane
    if (o0 < C_) {                       // ---- Q: scaled, transposed [N][C]
        const float scale = 0.0625f;     // 256^-0.5
        unsigned short* qb = qs_t + (size_t)b * N_ * C_;
#pragma unroll
        for (int r = 0; r < 8; ++r)
            qb[(size_t)col * C_ + (o0 + hi * 8 + r)] = f2bf(acc[r] * scale);
    } else if (o0 < 2 * C_) {            // ---- K: c-pair words [C/2][N]
        int oo = o0 - C_;
        uint32_t* kb = k_pk + (size_t)b * (C_ / 2) * N_;
#pragma unroll
        for (int j = 0; j < 4; ++j) {
            uint32_t wd = pack2bf(acc[2 * j], acc[2 * j + 1]);
            kb[(size_t)((oo + hi * 8 + 2 * j) >> 1) * N_ + col] = wd;
        }
    } else {                             // ---- V: n-pair words [N/2][C]
        int oo = o0 - 2 * C_;
        uint32_t* vb = v_pk + (size_t)b * (N_ / 2) * C_;
#pragma unroll
        for (int r = 0; r < 8; ++r) {
            float other = __shfl_xor(acc[r], 1, 32);   // neighbor column n^1
            if ((lane & 1) == 0) {
                uint32_t wd = pack2bf(acc[r], other);
                vb[(size_t)(col >> 1) * C_ + (oo + hi * 8 + r)] = wd;
            }
        }
    }
}

// ---------------------------------------------------------------------------
// Kernel 3: flash attention.  One wave per 16 query rows; streams key/value
// columns in 32-wide tiles with online softmax.  O accumulator: 16x256 f32
// in registers (16 x v8f).  P relayout D->A through per-wave LDS.
// ---------------------------------------------------------------------------
__global__ __launch_bounds__(128) void attn_kernel(
    const unsigned short* __restrict__ qs_t, const uint32_t* __restrict__ k_pk,
    const uint32_t* __restrict__ v_pk, unsigned short* __restrict__ att_t) {
    __shared__ unsigned short pbuf[4][16 * 32];   // per-wave P staging (1 KB each)
    int wid  = threadIdx.x >> 5;
    int lane = threadIdx.x & 31;
    int gw   = blockIdx.x * 4 + wid;              // 0..1023
    int nt   = gw % (N_ / 16);
    int b    = gw / (N_ / 16);
    int n0   = nt * 16;
    int row  = lane & 15;
    int hi   = lane >> 4;

    const unsigned short* qrow0 = qs_t + ((size_t)b * N_ + n0 + row) * C_;
    const uint32_t* kb = k_pk + (size_t)b * (C_ / 2) * N_;
    const uint32_t* vb = v_pk + (size_t)b * (N_ / 2) * C_;
    unsigned short* plds = pbuf[wid];

    v8f o_acc[16];
#pragma unroll
    for (int t = 0; t < 16; ++t)
        o_acc[t] = (v8f){0.f,0.f,0.f,0.f,0.f,0.f,0.f,0.f};
    float rmax[8], rsum[8];
#pragma unroll
    for (int r = 0; r < 8; ++r) { rmax[r] = -1e30f; rsum[r] = 0.f; }

    for (int m0 = 0; m0 < N_; m0 += 32) {
        // prefetch next key/value tile columns into cache (global_prefetch_b8)
        if (m0 + 32 < N_) {
            __builtin_prefetch(kb + (size_t)(hi * 8) * N_ + m0 + 32 + row, 0, 1);
            __builtin_prefetch(vb + (size_t)((m0 + 32) >> 1) * C_ + lane, 0, 1);
        }
        // ---- S = Qt[n0:16, :] @ K[:, m0:32]  (two 16x16 f32 tiles)
        v8f s0 = {0.f,0.f,0.f,0.f,0.f,0.f,0.f,0.f};
        v8f s1 = {0.f,0.f,0.f,0.f,0.f,0.f,0.f,0.f};
#pragma unroll
        for (int kk = 0; kk < 8; ++kk) {
            AF qa, B0, B1;
            const unsigned short* qp = qrow0 + kk * 32 + hi * 16;
            load16B(&qa.w[0], qp);
            load16B(&qa.w[4], qp + 8);
            int cb = kk * 16 + hi * 8;     // c-pair base
#pragma unroll
            for (int vv = 0; vv < 8; ++vv) {
                size_t rbase = (size_t)(cb + vv) * N_ + m0;
                B0.w[vv] = kb[rbase + row];
                B1.w[vv] = kb[rbase + 16 + row];
            }
            s0 = wmma_bf16(qa.v, B0.v, s0);
            s1 = wmma_bf16(qa.v, B1.v, s1);
        }
        // ---- online softmax (row reductions across the 16-lane half)
        float tmax[8];
#pragma unroll
        for (int r = 0; r < 8; ++r) tmax[r] = fmaxf(s0[r], s1[r]);
#pragma unroll
        for (int off = 8; off >= 1; off >>= 1)
#pragma unroll
            for (int r = 0; r < 8; ++r)
                tmax[r] = fmaxf(tmax[r], __shfl_xor(tmax[r], off, 32));
        float alpha[8], tsum[8];
#pragma unroll
        for (int r = 0; r < 8; ++r) {
            float nm = fmaxf(rmax[r], tmax[r]);
            alpha[r] = __expf(rmax[r] - nm);
            rmax[r]  = nm;
            s0[r] = __expf(s0[r] - nm);
            s1[r] = __expf(s1[r] - nm);
            tsum[r] = s0[r] + s1[r];
        }
#pragma unroll
        for (int off = 8; off >= 1; off >>= 1)
#pragma unroll
            for (int r = 0; r < 8; ++r)
                tsum[r] += __shfl_xor(tsum[r], off, 32);
#pragma unroll
        for (int r = 0; r < 8; ++r) rsum[r] = rsum[r] * alpha[r] + tsum[r];
#pragma unroll
        for (int t = 0; t < 16; ++t)
#pragma unroll
            for (int r = 0; r < 8; ++r)
                o_acc[t][r] *= alpha[r];

        // ---- P (D layout) -> LDS -> A fragment
#pragma unroll
        for (int r = 0; r < 8; ++r) {
            int rr = hi * 8 + r;
            plds[rr * 32 + row]      = f2bf(s0[r]);
            plds[rr * 32 + 16 + row] = f2bf(s1[r]);
        }
        asm volatile("s_wait_dscnt 0" ::: "memory");
        AF pf;
        const unsigned short* pp = plds + row * 32 + hi * 16;
        load16B(&pf.w[0], pp);
        load16B(&pf.w[4], pp + 8);
        asm volatile("s_wait_dscnt 0" ::: "memory");

        // ---- O += P @ V^T   (16 c-subtiles, K = 32 keys)
        int mbase = (m0 >> 1) + hi * 8;
#pragma unroll
        for (int t = 0; t < 16; ++t) {
            AF Bv;
#pragma unroll
            for (int vv = 0; vv < 8; ++vv)
                Bv.w[vv] = vb[(size_t)(mbase + vv) * C_ + t * 16 + row];
            o_acc[t] = wmma_bf16(pf.v, Bv.v, o_acc[t]);
        }
    }

    // ---- finalize: divide by row sums, store att_t [B][N][C] bf16
    unsigned short* ab = att_t + (size_t)b * N_ * C_;
#pragma unroll
    for (int t = 0; t < 16; ++t) {
#pragma unroll
        for (int r = 0; r < 8; ++r) {
            int rr = hi * 8 + r;
            float o = o_acc[t][r] / rsum[r];
            ab[(size_t)(n0 + rr) * C_ + t * 16 + row] = f2bf(o);
        }
    }
}

// ---------------------------------------------------------------------------
// Kernel 4: proj GEMM + bias + residual, f32 output [B][C][N].
// ---------------------------------------------------------------------------
__global__ __launch_bounds__(256) void proj_kernel(
    const unsigned short* __restrict__ att_t, const unsigned short* __restrict__ wp_bf,
    const float* __restrict__ proj_b, const float* __restrict__ x,
    float* __restrict__ out) {
    int wave = (int)((blockIdx.x * blockDim.x + threadIdx.x) >> 5);
    int lane = threadIdx.x & 31;
    int nt = wave % (N_ / 16);
    int ot = (wave / (N_ / 16)) % (C_ / 16);
    int b  = wave / ((N_ / 16) * (C_ / 16));
    int o0 = ot * 16, n0 = nt * 16;
    int row = lane & 15;
    int hi  = lane >> 4;

    const unsigned short* ab = att_t + (size_t)b * N_ * C_;
    const uint32_t* arow = (const uint32_t*)(ab + (size_t)(n0 + row) * C_);
    const unsigned short* wrow = wp_bf + (size_t)(o0 + row) * C_;
    v8f acc = {0.f,0.f,0.f,0.f,0.f,0.f,0.f,0.f};

    for (int c0 = 0; c0 < C_; c0 += 32) {
        AF A, Bf;
        const unsigned short* wp = wrow + c0 + hi * 16;
        load16B(&A.w[0], wp);
        load16B(&A.w[4], wp + 8);
#pragma unroll
        for (int vv = 0; vv < 8; ++vv)
            Bf.w[vv] = arow[(c0 >> 1) + hi * 8 + vv];
        acc = wmma_bf16(A.v, Bf.v, acc);
    }
    const float* xb = x   + (size_t)b * C_ * N_;
    float*       ob = out + (size_t)b * C_ * N_;
    int col = n0 + row;
#pragma unroll
    for (int r = 0; r < 8; ++r) {
        int o = o0 + hi * 8 + r;
        ob[(size_t)o * N_ + col] = acc[r] + proj_b[o] + xb[(size_t)o * N_ + col];
    }
}

// ---------------------------------------------------------------------------
extern "C" void kernel_launch(void* const* d_in, const int* in_sizes, int n_in,
                              void* d_out, int out_size, void* d_ws, size_t ws_size,
                              hipStream_t stream) {
    const float* x      = (const float*)d_in[0];
    const float* norm_w = (const float*)d_in[1];
    const float* norm_b = (const float*)d_in[2];
    const float* qkv_w  = (const float*)d_in[3];
    const float* qkv_b  = (const float*)d_in[4];
    const float* proj_w = (const float*)d_in[5];
    const float* proj_b = (const float*)d_in[6];
    float* out = (float*)d_out;

    char* ws = (char*)d_ws;
    const size_t szPk   = (size_t)B_ * (C_ / 2) * N_ * 4;    // 8 MB (dword pairs)
    const size_t szHalf = (size_t)B_ * N_ * C_ * 2;          // 8 MB (bf16 plane)
    const size_t szWq   = (size_t)OQKV * C_ * 2;             // 384 KB
    uint32_t*       h_pk  = (uint32_t*)(ws);
    unsigned short* qs_t  = (unsigned short*)(ws + szPk);
    uint32_t*       k_pk  = (uint32_t*)(ws + szPk + szHalf);
    uint32_t*       v_pk  = (uint32_t*)(ws + 2 * szPk + szHalf);
    unsigned short* att_t = (unsigned short*)(ws + 3 * szPk + szHalf);
    unsigned short* wq_bf = (unsigned short*)(ws + 3 * szPk + 2 * szHalf);
    unsigned short* wp_bf = (unsigned short*)(ws + 3 * szPk + 2 * szHalf + szWq);

    // one-shot weight conversion (cheap; overlaps nothing but is tiny)
    const int convWords = (OQKV * C_ + C_ * C_) / 2;          // 131072
    wconv_kernel<<<(convWords + 255) / 256, 256, 0, stream>>>(
        qkv_w, proj_w, (uint32_t*)wq_bf, (uint32_t*)wp_bf);

    gn_kernel<<<B_ * G_, 256, 0, stream>>>(x, norm_w, norm_b, h_pk);

    int qkv_waves = B_ * (OQKV / 16) * (N_ / 16);            // 49152
    qkv_kernel<<<qkv_waves / 8, 256, 0, stream>>>(h_pk, wq_bf, qkv_b,
                                                  qs_t, k_pk, v_pk);

    int attn_waves = B_ * (N_ / 16);                          // 1024
    attn_kernel<<<attn_waves / 4, 128, 0, stream>>>(qs_t, k_pk, v_pk, att_t);

    int proj_waves = B_ * (C_ / 16) * (N_ / 16);              // 16384
    proj_kernel<<<proj_waves / 8, 256, 0, stream>>>(att_t, wp_bf, proj_b, x, out);
}
